// MoE_14903536517182
// MI455X (gfx1250) — compile-verified, matrix operand order checked by reading, not run on hardware
//
#include <hip/hip_runtime.h>
#include <cstdint>

#define NTOK 8192      // B*T
#define CDIM 768
#define HDIM 3072
#define NEXP 8
#define MTILE 48       // tokens per block (3 WMMA M-tiles); keeps VGPRs < 256
#define NMT 3          // M-tiles per block
#define XPAD 8         // LDS row padding (halves) -> conflict-free b128 lanes
#define HPAD 8
#define JSTRIDE 98304  // byte stride between adjacent 16-wide n-tiles of W2^T

typedef __attribute__((ext_vector_type(16))) __bf16 v16bf;
typedef __attribute__((ext_vector_type(8)))  float  v8f;
typedef __attribute__((ext_vector_type(4)))  uint   v4u;

union V16 { v16bf v; v4u u[2]; };

// ---------------------------------------------------------------------------
// WMMA helper (CDNA5 16x16x32 bf16, f32 accumulate)
// ---------------------------------------------------------------------------
__device__ __forceinline__ v8f wmma_bf16(v16bf a, v16bf b, v8f c) {
  // 8 args: (neg_a, A, neg_b, B, c_mod, C, reuse_a, reuse_b)
  return __builtin_amdgcn_wmma_f32_16x16x32_bf16(false, a, false, b, (short)0, c,
                                                 false, false);
}

// Hand-issued B-tile load: two b128 chunks per lane (ISA 16-bit A/B striping:
// halves {k..k+7} at +0 and {k+16..k+23} at +32B). Volatile asm preserves the
// issue order of the software pipeline; the compiler does not model these as
// VMEM, so waitcnt discipline below is manual.
__device__ __forceinline__ void tile_load(V16& d, const __bf16* base, int off) {
  asm volatile("global_load_b128 %0, %2, off offset:%c3\n\t"
               "global_load_b128 %1, %2, off offset:%c4"
               : "=&v"(d.u[0]), "=&v"(d.u[1])
               : "v"(base), "i"(off), "i"(off + 32));
}

// Manual wait: block until <= n VMEM loads outstanding (loads complete in
// order, so n = 2 loads/tile * tiles-allowed-in-flight). Tying the consumed
// tile's registers as "+v" operands orders the consuming WMMAs after the wait.
__device__ __forceinline__ void tile_wait(V16& d, int n) {
  asm volatile("s_wait_loadcnt %c2"
               : "+v"(d.u[0]), "+v"(d.u[1])
               : "i"(n));
}

// Compiler-managed LDS A-tile load (tracked by DScnt as usual).
__device__ __forceinline__ v16bf load_tile_lds(const __bf16* p) {
  V16 t;
  t.u[0] = *(const v4u*)p;
  t.u[1] = *(const v4u*)(p + 16);
  return t.v;
}

// ---------------------------------------------------------------------------
// x -> bf16 (streaming)
// ---------------------------------------------------------------------------
__global__ __launch_bounds__(256) void cvt_x_kernel(const float* __restrict__ x,
                                                    __bf16* __restrict__ xb) {
  int i = blockIdx.x * 256 + threadIdx.x;
  if (i < NTOK * CDIM) xb[i] = (__bf16)x[i];
}

// ---------------------------------------------------------------------------
// Tiled transpose + f32->bf16 convert: dst[e][l][r] = bf16(src[e][r][l]).
// Coalesced on both sides via 32x32 LDS tile. block (32,8), grid (L/32, R/32, E).
// ---------------------------------------------------------------------------
__global__ __launch_bounds__(256) void transpose_cvt_kernel(
    const float* __restrict__ src, __bf16* __restrict__ dst, int R, int L) {
  __shared__ float tile[32][33];
  const int e  = blockIdx.z;
  const int l0 = blockIdx.x * 32;
  const int r0 = blockIdx.y * 32;
  const float* s = src + (size_t)e * R * L;
  __bf16*       d = dst + (size_t)e * L * R;
  const int tx = threadIdx.x, ty = threadIdx.y;
#pragma unroll
  for (int k = 0; k < 32; k += 8)
    tile[ty + k][tx] = s[(size_t)(r0 + ty + k) * L + (l0 + tx)];
  __syncthreads();
#pragma unroll
  for (int k = 0; k < 32; k += 8)
    d[(size_t)(l0 + ty + k) * R + (r0 + tx)] = (__bf16)tile[tx][ty + k];
}

// ---------------------------------------------------------------------------
// Gating: scores = x @ Wg^T, top-2 (lower index wins ties, like lax.top_k),
// 2-way softmax, append (token, weight) to per-expert routing lists.
// ---------------------------------------------------------------------------
__global__ __launch_bounds__(256) void gate_kernel(const float* __restrict__ x,
                                                   const float* __restrict__ Wg,
                                                   int* __restrict__ tok,
                                                   float* __restrict__ gwb,
                                                   int* __restrict__ cnt,
                                                   int2* __restrict__ gsel,
                                                   float2* __restrict__ gwsel) {
  __shared__ float WgS[NEXP][CDIM];
  const int tid = threadIdx.x;
  for (int i = tid; i < NEXP * CDIM; i += 256) WgS[i / CDIM][i % CDIM] = Wg[i];
  __syncthreads();

  const int lane = tid & 31;
  const int wave = tid >> 5;
  const int n = blockIdx.x * 8 + wave;   // one wave per token

  float p[NEXP];
#pragma unroll
  for (int e = 0; e < NEXP; ++e) p[e] = 0.0f;
  const float* xr = x + (size_t)n * CDIM;
  for (int c = lane; c < CDIM; c += 32) {
    float xv = xr[c];
#pragma unroll
    for (int e = 0; e < NEXP; ++e) p[e] += xv * WgS[e][c];
  }
#pragma unroll
  for (int e = 0; e < NEXP; ++e)
    for (int off = 16; off > 0; off >>= 1) p[e] += __shfl_xor(p[e], off, 32);

  if (lane == 0) {
    int e0 = 0;
    for (int e = 1; e < NEXP; ++e) if (p[e] > p[e0]) e0 = e;   // strict >: lower idx wins
    int e1 = (e0 == 0) ? 1 : 0;
    for (int e = 0; e < NEXP; ++e) if (e != e0 && p[e] > p[e1]) e1 = e;
    const float mx = fmaxf(p[e0], p[e1]);
    const float x0 = expf(p[e0] - mx), x1 = expf(p[e1] - mx);
    const float inv = 1.0f / (x0 + x1);
    const float w0 = x0 * inv, w1 = x1 * inv;
    int p0 = atomicAdd(&cnt[e0], 1);
    tok[e0 * NTOK + p0] = n;  gwb[e0 * NTOK + p0] = w0;
    int p1 = atomicAdd(&cnt[e1], 1);
    tok[e1 * NTOK + p1] = n;  gwb[e1 * NTOK + p1] = w1;
    gsel[n]  = make_int2(e0, e1);
    gwsel[n] = make_float2(w0, w1);
  }
}

// ---------------------------------------------------------------------------
// Aux loss: deterministic fixed-order reduction of per-token gate weights.
// ---------------------------------------------------------------------------
__global__ __launch_bounds__(256) void aux_kernel(const int2* __restrict__ gsel,
                                                  const float2* __restrict__ gwsel,
                                                  float* __restrict__ aux_out) {
  __shared__ float red[256];
  __shared__ float usage[NEXP];
  const int tid = threadIdx.x;
  float loc[NEXP];
#pragma unroll
  for (int e = 0; e < NEXP; ++e) loc[e] = 0.0f;
  for (int n = tid; n < NTOK; n += 256) {
    int2 ee = gsel[n]; float2 ww = gwsel[n];
    loc[ee.x] += ww.x;
    loc[ee.y] += ww.y;
  }
  for (int e = 0; e < NEXP; ++e) {
    red[tid] = loc[e];
    __syncthreads();
    for (int s = 128; s > 0; s >>= 1) {
      if (tid < s) red[tid] += red[tid + s];
      __syncthreads();
    }
    if (tid == 0) usage[e] = red[0] / (float)NTOK;
    __syncthreads();
  }
  if (tid == 0) {
    float ent = 0.0f;
    for (int e = 0; e < NEXP; ++e) {
      float u = usage[e];
      ent -= u * logf(u + 1e-10f);
    }
    aux_out[0] = 0.01f * (1.0f - ent / logf((float)NEXP));
  }
}

// ---------------------------------------------------------------------------
// Fused expert MLP, M=48 token tile per block (8 waves, 256 threads).
// B tiles: hand-issued asm global_load pairs in a 3-slot rotation, distance-2
// prefetch, with manual s_wait_loadcnt tied to the consumed tile -> each tile
// is in flight across two 3-WMMA groups; issue order is literal (volatile).
// ---------------------------------------------------------------------------
__global__ __launch_bounds__(256) void moe_expert_kernel(
    const __bf16* __restrict__ xb, const __bf16* __restrict__ w1t,
    const __bf16* __restrict__ w2t, const float* __restrict__ b1,
    const float* __restrict__ b2, const int* __restrict__ tok,
    const float* __restrict__ gwb, const int* __restrict__ cnt,
    float* __restrict__ y) {
  const int e = blockIdx.y;
  const int cn = cnt[e];
  const int row0 = blockIdx.x * MTILE;
  if (row0 >= cn) return;                    // uniform early-exit
  const int nrows = min(MTILE, cn - row0);

  __shared__ __align__(16) __bf16 Xt[MTILE][CDIM + XPAD];   // ~73 KB
  __shared__ __align__(16) __bf16 Ht[MTILE][128 + HPAD];    // ~13 KB
  __shared__ int   tokS[MTILE];
  __shared__ float gwS[MTILE];

  const int tid = threadIdx.x;
  const int lane = tid & 31;
  const int wave = tid >> 5;
  const int m8  = (lane >> 4) * 8;      // C/D rows held by this lane
  const int n16 = lane & 15;            // C/D column within a 16-wide tile
  const int kb  = (lane >> 4) << 3;     // per-lane K sub-offset of A/B striping

  if (tid < MTILE) {
    const bool valid = tid < nrows;
    tokS[tid] = valid ? tok[e * NTOK + row0 + tid] : 0;
    gwS[tid]  = valid ? gwb[e * NTOK + row0 + tid] : 0.0f;
  }
  __syncthreads();

  // Gather the 48 token rows (bf16) into LDS, 16B chunks, coalesced per row.
  for (int i = tid; i < MTILE * (CDIM / 8); i += 256) {
    const int r  = i / (CDIM / 8);
    const int cc = (i - r * (CDIM / 8)) * 8;
    const v4u v = *(const v4u*)(xb + (size_t)tokS[r] * CDIM + cc);
    *(v4u*)&Xt[r][cc] = v;
  }
  __syncthreads();

  const __bf16* W1e = w1t + (size_t)e * HDIM * CDIM;  // [h][c]
  const __bf16* W2e = w2t + (size_t)e * CDIM * HDIM;  // [c][h]

  // Per-lane LDS A-tile bases (static ds offsets stay within the 16-bit field).
  const __bf16* xtp[NMT];
  const __bf16* htp[NMT];
#pragma unroll
  for (int mt = 0; mt < NMT; ++mt) {
    xtp[mt] = &Xt[mt * 16 + n16][kb];
    htp[mt] = &Ht[mt * 16 + n16][kb];
  }

  // Single per-lane weight pointers; all tile offsets inside a chunk are
  // compile-time immediates (j-stride 96KB, kc-stride 64B < 24-bit ioffset).
  const __bf16* w1p = W1e + (size_t)(wave * 16 + n16) * CDIM + kb;
  const __bf16* w2p = W2e + (size_t)(wave * 96 + n16) * HDIM + kb;

  const v8f vzero = {0.f, 0.f, 0.f, 0.f, 0.f, 0.f, 0.f, 0.f};
  v8f acc2[NMT][6];
#pragma unroll
  for (int mt = 0; mt < NMT; ++mt)
#pragma unroll
    for (int j = 0; j < 6; ++j) acc2[mt][j] = vzero;

  for (int hb = 0; hb < HDIM; hb += 128) {
    // ---- Phase A: h-columns [hb+16*wave, +16) for all 3 M-tiles. ----
    v8f acc1[NMT];
#pragma unroll
    for (int mt = 0; mt < NMT; ++mt) acc1[mt] = vzero;

    V16 bA[3];
    tile_load(bA[0], w1p, 0);
    tile_load(bA[1], w1p, 64);
#pragma unroll
    for (int s = 0; s < CDIM / 32; ++s) {      // 24 steps, kc = 32*s
      if (s + 2 < CDIM / 32) tile_load(bA[(s + 2) % 3], w1p, 64 * (s + 2));
      tile_wait(bA[s % 3], (s + 2 < CDIM / 32) ? 4 : 2 * (CDIM / 32 - 1 - s));
#pragma unroll
      for (int mt = 0; mt < NMT; ++mt) {
        v16bf a = load_tile_lds(xtp[mt] + 32 * s);
        acc1[mt] = wmma_bf16(a, bA[s % 3].v, acc1[mt]);
      }
    }

    const int hcol = hb + wave * 16;
    const float bias1 = b1[e * HDIM + hcol + n16];
    const int hl = wave * 16 + n16;
#pragma unroll
    for (int mt = 0; mt < NMT; ++mt) {
#pragma unroll
      for (int i = 0; i < 8; ++i) {
        float v = acc1[mt][i] + bias1;
        v = 0.5f * v * (1.0f + erff(v * 0.70710678118654752f));  // exact GELU
        Ht[mt * 16 + m8 + i][hl] = (__bf16)v;
      }
    }
    __syncthreads();

    // ---- Phase B: acc2 += Htile @ W2[hb:hb+128, :], 24-step pipeline. ----
    v16bf a[NMT];
    V16 bB[3];
    tile_load(bB[0], w2p, 0);
    tile_load(bB[1], w2p, JSTRIDE);
#pragma unroll
    for (int t = 0; t < 24; ++t) {
      const int kc = (t / 6) * 32;
      const int j  = t % 6;
      if (j == 0) {
#pragma unroll
        for (int mt = 0; mt < NMT; ++mt) a[mt] = load_tile_lds(htp[mt] + kc);
      }
      const int tn = t + 2;
      if (tn < 24) tile_load(bB[tn % 3], w2p, (tn % 6) * JSTRIDE + (tn / 6) * 64);
      tile_wait(bB[t % 3], (tn < 24) ? 4 : 2 * (23 - t));
#pragma unroll
      for (int mt = 0; mt < NMT; ++mt)
        acc2[mt][j] = wmma_bf16(a[mt], bB[t % 3].v, acc2[mt][j]);
    }
    __syncthreads();  // Ht reused next chunk

    w1p += (size_t)128 * CDIM;   // next 128 h-rows of W1^T
    w2p += 128;                  // next 128 K of W2^T rows
  }

  // ---- Epilogue: +b2, x gate weight, deterministic f32 atomic combine. ----
#pragma unroll
  for (int j = 0; j < 6; ++j) {
    const int n = (wave * 6 + j) * 16 + n16;
    const float bias2 = b2[e * CDIM + n];
#pragma unroll
    for (int mt = 0; mt < NMT; ++mt) {
#pragma unroll
      for (int i = 0; i < 8; ++i) {
        const int m = mt * 16 + m8 + i;
        if (m < nrows) {
          const float val = (acc2[mt][j][i] + bias2) * gwS[m];
          atomicAdd(&y[(size_t)tokS[m] * CDIM + n], val);
        }
      }
    }
  }
}

// ---------------------------------------------------------------------------
extern "C" void kernel_launch(void* const* d_in, const int* in_sizes, int n_in,
                              void* d_out, int out_size, void* d_ws, size_t ws_size,
                              hipStream_t stream) {
  const float* x  = (const float*)d_in[0];
  const float* Wg = (const float*)d_in[1];
  const float* W1 = (const float*)d_in[2];
  const float* b1 = (const float*)d_in[3];
  const float* W2 = (const float*)d_in[4];
  const float* b2 = (const float*)d_in[5];
  float* out = (float*)d_out;

  // Workspace carve-up (256B aligned): ~89 MB total.
  char* ws = (char*)d_ws;
  size_t off = 0;
  auto carve = [&](size_t bytes) -> void* {
    off = (off + 255) & ~(size_t)255;
    void* p = ws + off;
    off += bytes;
    return p;
  };
  int*    cnt   = (int*)   carve(NEXP * sizeof(int));
  int*    tok   = (int*)   carve((size_t)NEXP * NTOK * sizeof(int));
  float*  gwb   = (float*) carve((size_t)NEXP * NTOK * sizeof(float));
  int2*   gsel  = (int2*)  carve((size_t)NTOK * sizeof(int2));
  float2* gwsel = (float2*)carve((size_t)NTOK * sizeof(float2));
  __bf16* xb    = (__bf16*)carve((size_t)NTOK * CDIM * sizeof(__bf16));
  __bf16* w1t   = (__bf16*)carve((size_t)NEXP * HDIM * CDIM * sizeof(__bf16));
  __bf16* w2t   = (__bf16*)carve((size_t)NEXP * CDIM * HDIM * sizeof(__bf16));

  hipMemsetAsync(out, 0, (size_t)out_size * sizeof(float), stream);
  hipMemsetAsync(cnt, 0, NEXP * sizeof(int), stream);

  cvt_x_kernel<<<(NTOK * CDIM + 255) / 256, 256, 0, stream>>>(x, xb);
  // w1t[e][h][c] = bf16(W1[e][c][h]) : R=CDIM rows, L=HDIM cols
  transpose_cvt_kernel<<<dim3(HDIM / 32, CDIM / 32, NEXP), dim3(32, 8), 0, stream>>>(
      W1, w1t, CDIM, HDIM);
  // w2t[e][c][h] = bf16(W2[e][h][c]) : R=HDIM rows, L=CDIM cols
  transpose_cvt_kernel<<<dim3(CDIM / 32, HDIM / 32, NEXP), dim3(32, 8), 0, stream>>>(
      W2, w2t, HDIM, CDIM);

  gate_kernel<<<NTOK / 8, 256, 0, stream>>>(x, Wg, tok, gwb, cnt, gsel, gwsel);
  aux_kernel<<<1, 256, 0, stream>>>(gsel, gwsel, out + (size_t)NTOK * CDIM);

  moe_expert_kernel<<<dim3((NTOK + MTILE - 1) / MTILE, NEXP), 256, 0, stream>>>(
      xb, w1t, w2t, b1, b2, tok, gwb, cnt, out);
}
